// DogDetector_87136296501853
// MI455X (gfx1250) — matrix operand order repaired and checked by reading, not compiled, not run on hardware
//
#include <hip/hip_runtime.h>
#include <hip/hip_bf16.h>
#include <math.h>

typedef __attribute__((ext_vector_type(16))) _Float16 v16h;
typedef __attribute__((ext_vector_type(8)))  float    v8f;
typedef unsigned int v4u __attribute__((ext_vector_type(4)));
typedef int          v8i __attribute__((ext_vector_type(8)));
typedef int          v4i __attribute__((ext_vector_type(4)));

#if __has_builtin(__builtin_amdgcn_tensor_load_to_lds) && __has_builtin(__builtin_amdgcn_s_wait_tensorcnt)
#define HAVE_TDM 1
#else
#define HAVE_TDM 0
#endif

// ---------------------------------------------------------------------------
// Epilogue flags for the fused implicit-GEMM conv
// ---------------------------------------------------------------------------
#define EPI_RELU    1
#define EPI_SIGMOID 2
#define EPI_BIAS    4
#define EPI_BN      8
#define EPI_RES     16

#define BN_EPS 1e-5f

#if HAVE_TDM
// Issue a TDM load of a [rows<=64] x [32-col] f32 tile (row stride = rowStride
// elements) from global into LDS at byte offset lds_off. Hardware zero-fills
// reads past tensor_dim0 (=colsRem) / tensor_dim1 (=rowsRem).
__device__ __forceinline__ void tdm_load_w_tile(const float* gptr, unsigned lds_off,
                                                unsigned rowsRem, unsigned colsRem,
                                                unsigned rowStride)
{
  unsigned long long ga = (unsigned long long)(uintptr_t)gptr;
  v4u g0;
  g0[0] = 1u;                                            // count=1 (user D#)
  g0[1] = lds_off;                                       // lds_addr (bytes)
  g0[2] = (unsigned)(ga & 0xffffffffull);                // global_addr[31:0]
  g0[3] = (unsigned)((ga >> 32) & 0x1ffffffull) | (2u << 30); // addr[56:32], type=2
  v8i g1;
  unsigned long long st0 = (unsigned long long)rowStride; // tensor_dim0_stride
  g1[0] = (int)(2u << 16);                               // data_size = 4 bytes
  g1[1] = (int)((colsRem & 0xffffu) << 16);              // tensor_dim0 lo16 @ bits 63:48
  g1[2] = (int)(((colsRem >> 16) & 0xffffu) | ((rowsRem & 0xffffu) << 16)); // td0 hi / td1 lo
  g1[3] = (int)(((rowsRem >> 16) & 0xffffu) | (32u << 16)); // td1 hi, tile_dim0=32
  g1[4] = (int)64u;                                      // tile_dim1=64, tile_dim2=0
  g1[5] = (int)(st0 & 0xffffffffull);
  g1[6] = (int)((st0 >> 32) & 0xffffull);                // stride hi, td1_stride lo = 0
  g1[7] = 0;
  v4i z4 = {0, 0, 0, 0};
  v8i z8 = {0, 0, 0, 0, 0, 0, 0, 0};
  __builtin_amdgcn_tensor_load_to_lds(g0, g1, z4, z4, z8, 0);
}
#endif

// ---------------------------------------------------------------------------
// Implicit-GEMM convolution on v_wmma_f32_16x16x32_f16.
//   D[Cout, N*Ho*Wo] = W[Cout, Cin*KH*KW] x im2col(X)[Cin*KH*KW, N*Ho*Wo]
// Block = 8 waves. blockIdx.y -> 64-row M supertile (shared weight tile),
// wave -> 16-column N tile. Each wave: 4 accumulators (64x16 output), one
// im2col B-gather per 32-wide K chunk feeds 4 WMMAs. Weight tile is staged
// into LDS by the Tensor Data Mover, double-buffered across K chunks.
// ---------------------------------------------------------------------------
__global__ __launch_bounds__(256) void conv_wmma_f16(
    const float* __restrict__ X, const float* __restrict__ Wt,
    const float* __restrict__ bias,
    const float* __restrict__ bn_g, const float* __restrict__ bn_b,
    const float* __restrict__ bn_m, const float* __restrict__ bn_v,
    const float* __restrict__ res, float* __restrict__ Y,
    int Nimg, int Cin, int Hin, int Win,
    int Cout, int Hout, int Wout,
    int KH, int KW, int stride, int pad, int flags)
{
  __shared__ float Atile[2][64 * 32];   // double-buffered 64x32 f32 weight tile

  const int lane = threadIdx.x & 31;
  const int wid  = threadIdx.x >> 5;
  const int Ntot = Nimg * Hout * Wout;
  const int m0   = blockIdx.y << 6;                 // 64-row M supertile
  const int n0   = (blockIdx.x * 8 + wid) << 4;     // 16-col N tile
  const bool active = (n0 < Ntot);
  const int K   = Cin * KH * KW;
  const int HW  = Hout * Wout;
  const int khw = KH * KW;
  const int nchunks = (K + 31) >> 5;

  // Wave-uniform decomposition of this wave's 16 output columns.
  int nimg[16], noy[16], nox[16];
  bool nval[16];
#pragma unroll
  for (int e = 0; e < 16; ++e) {
    int n = n0 + e;
    if (active && n < Ntot) {
      nval[e] = true;
      int im = n / HW; int r = n - im * HW;
      nimg[e] = im; noy[e] = r / Wout; nox[e] = r - (r / Wout) * Wout;
    } else { nval[e] = false; nimg[e] = 0; noy[e] = 0; nox[e] = 0; }
  }

  v8f acc0, acc1, acc2, acc3;
#pragma unroll
  for (int r = 0; r < 8; ++r) { acc0[r] = 0.f; acc1[r] = 0.f; acc2[r] = 0.f; acc3[r] = 0.f; }

  const int khalf8 = ((lane >> 4) & 1) * 8;
  const int rowA   = lane & 15;

#if HAVE_TDM
  // Prologue: stage chunk 0.
  if (wid == 0 && lane == 0)
    tdm_load_w_tile(Wt + (long long)m0 * K, (unsigned)(uintptr_t)&Atile[0][0],
                    (unsigned)(Cout - m0), (unsigned)K, (unsigned)K);
#endif

  for (int kc = 0; kc < nchunks; ++kc) {
    const int k0  = kc << 5;
    const int buf = kc & 1;

#if HAVE_TDM
    if (wid == 0 && lane == 0) {
      if (kc + 1 < nchunks) {
        tdm_load_w_tile(Wt + (long long)m0 * K + (k0 + 32),
                        (unsigned)(uintptr_t)&Atile[buf ^ 1][0],
                        (unsigned)(Cout - m0), (unsigned)(K - (k0 + 32)), (unsigned)K);
        __builtin_amdgcn_s_wait_tensorcnt(1);   // current chunk complete (in-order)
      } else {
        __builtin_amdgcn_s_wait_tensorcnt(0);
      }
    }
    __syncthreads();                            // weight tile visible to all waves
#endif

    if (active) {
      // B fragment: lane owns im2col column n0+(lane&15), 16 k rows in table order.
      v16h b;
#pragma unroll
      for (int e = 0; e < 16; ++e) {
        int p = e >> 1, h = e & 1;
        const int kb = k0 + ((p & 3) << 1) + h + ((p >> 2) << 4) + khalf8;
        float v = 0.0f;
        if (kb < K && nval[rowA]) {
          int ci = kb / khw; int rk = kb - ci * khw;
          int ky = rk / KW;  int kx = rk - ky * KW;
          int iy = noy[rowA] * stride - pad + ky;
          int ix = nox[rowA] * stride - pad + kx;
          if (iy >= 0 && iy < Hin && ix >= 0 && ix < Win)
            v = X[(((long long)nimg[rowA] * Cin + ci) * Hin + iy) * Win + ix];
        }
        b[e] = (_Float16)v;
      }

      // 4 M-subtiles share the B fragment.
#pragma unroll
      for (int msub = 0; msub < 4; ++msub) {
        if (m0 + (msub << 4) < Cout) {          // wave-uniform guard
          v16h a;
#pragma unroll
          for (int e = 0; e < 16; ++e) {
            int p = e >> 1, h = e & 1;
            int koff = ((p & 3) << 1) + h + ((p >> 2) << 4) + khalf8;
#if HAVE_TDM
            float v = Atile[buf][(((msub << 4) + rowA) << 5) + koff];  // ds_load
#else
            int mA = m0 + (msub << 4) + rowA;
            int k  = k0 + koff;
            float v = (mA < Cout && k < K) ? Wt[(long long)mA * K + k] : 0.0f;
#endif
            a[e] = (_Float16)v;
          }
          v8f* accp = (msub == 0) ? &acc0 : (msub == 1) ? &acc1 : (msub == 2) ? &acc2 : &acc3;
          *accp = __builtin_amdgcn_wmma_f32_16x16x32_f16(
              false, a, false, b, (short)0, *accp, false, false);
        }
      }
    }
#if HAVE_TDM
    __syncthreads();                            // all reads done before next DMA overwrites
#endif
  }

  // Epilogue: bias -> BN -> residual -> relu/sigmoid, scatter to NCHW.
  if (active) {
    const int nC = n0 + rowA;
    if (nC < Ntot) {
      int im = nC / HW; int rr = nC - im * HW;
#pragma unroll
      for (int msub = 0; msub < 4; ++msub) {
        const v8f* accp = (msub == 0) ? &acc0 : (msub == 1) ? &acc1 : (msub == 2) ? &acc2 : &acc3;
#pragma unroll
        for (int r = 0; r < 8; ++r) {
          int m = m0 + (msub << 4) + r + ((lane >> 4) & 1) * 8;
          if (m < Cout) {
            float v = (*accp)[r];
            if (flags & EPI_BIAS) v += bias[m];
            if (flags & EPI_BN) {
              float sc = bn_g[m] * rsqrtf(bn_v[m] + BN_EPS);
              v = v * sc + (bn_b[m] - bn_m[m] * sc);
            }
            long long oidx = ((long long)im * Cout + m) * HW + rr;
            if (flags & EPI_RES)  v += res[oidx];
            if (flags & EPI_RELU) v = fmaxf(v, 0.0f);
            if (flags & EPI_SIGMOID) v = 1.0f / (1.0f + __expf(-v));
            Y[oidx] = v;
          }
        }
      }
    }
  }
}

// ---------------------------------------------------------------------------
// Max pooling (NCHW), -inf padding semantics like lax.reduce_window.
// ---------------------------------------------------------------------------
__global__ void maxpool_kernel(const float* __restrict__ X, float* __restrict__ Y,
                               int N, int C, int Hi, int Wi, int Ho, int Wo,
                               int win, int stride, int pad)
{
  long long idx = (long long)blockIdx.x * blockDim.x + threadIdx.x;
  long long tot = (long long)N * C * Ho * Wo;
  if (idx >= tot) return;
  int x = (int)(idx % Wo); long long r = idx / Wo;
  int y = (int)(r % Ho); r /= Ho;
  int c = (int)(r % C);  int n = (int)(r / C);
  float m = -INFINITY;
  for (int ky = 0; ky < win; ++ky) {
    int iy = y * stride - pad + ky;
    if (iy < 0 || iy >= Hi) continue;
    for (int kx = 0; kx < win; ++kx) {
      int ix = x * stride - pad + kx;
      if (ix < 0 || ix >= Wi) continue;
      float v = X[(((long long)n * C + c) * Hi + iy) * Wi + ix];
      m = fmaxf(m, v);
    }
  }
  Y[idx] = m;
}

// Y[n,c,y,x] += S[n,c, y*Hs/Ho, x*Ws/Wo]   (nearest upsample + add)
__global__ void upsample_add_kernel(float* __restrict__ Y, const float* __restrict__ S,
                                    int N, int C, int Ho, int Wo, int Hs, int Ws)
{
  long long idx = (long long)blockIdx.x * blockDim.x + threadIdx.x;
  long long tot = (long long)N * C * Ho * Wo;
  if (idx >= tot) return;
  int x = (int)(idx % Wo); long long r = idx / Wo;
  int y = (int)(r % Ho); r /= Ho;
  int c = (int)(r % C);  int n = (int)(r / C);
  int sy = (int)(((long long)y * Hs) / Ho);
  int sx = (int)(((long long)x * Ws) / Wo);
  Y[idx] += S[(((long long)n * C + c) * Hs + sy) * Ws + sx];
}

__global__ void add_kernel(const float* __restrict__ A, const float* __restrict__ B,
                           float* __restrict__ Y, long long n)
{
  long long idx = (long long)blockIdx.x * blockDim.x + threadIdx.x;
  if (idx < n) Y[idx] = A[idx] + B[idx];
}

// ---------------------------------------------------------------------------
// Anchor decode: bbox [B,36,7,7], conf [B,9,7,7] (sigmoid already applied)
//   -> boxes [B,441,4], scores [B,441]
// ---------------------------------------------------------------------------
#define FMS 7
#define NA  9
#define NDET 441
__global__ void decode_kernel(const float* __restrict__ bbox, const float* __restrict__ conf,
                              float* __restrict__ boxes, float* __restrict__ scores, int B)
{
  int idx = blockIdx.x * blockDim.x + threadIdx.x;
  if (idx >= B * NDET) return;
  int b = idx / NDET, n = idx % NDET;
  int cell = n / NA, a = n % NA;
  int i = cell / FMS, j = cell % FMS;
  const float scales[3] = {0.25f, 0.5f, 0.75f};
  const float ratios[3] = {0.5f, 1.0f, 2.0f};
  float s  = scales[a / 3];
  float rr = ratios[a % 3];
  float aw = s * sqrtf(rr), ah = s / sqrtf(rr);
  float cx = (j + 0.5f) / (float)FMS, cy = (i + 0.5f) / (float)FMS;
  int sp = i * FMS + j;
  float t0 = bbox[((long long)b * 36 + (a * 4 + 0)) * 49 + sp];
  float t1 = bbox[((long long)b * 36 + (a * 4 + 1)) * 49 + sp];
  float t2 = bbox[((long long)b * 36 + (a * 4 + 2)) * 49 + sp];
  float t3 = bbox[((long long)b * 36 + (a * 4 + 3)) * 49 + sp];
  float pcx = t0 * aw + cx, pcy = t1 * ah + cy;
  float psx = __expf(t2) * aw, psy = __expf(t3) * ah;
  long long o = ((long long)b * NDET + n) * 4;
  boxes[o + 0] = pcx - psx * 0.5f;
  boxes[o + 1] = pcy - psy * 0.5f;
  boxes[o + 2] = pcx + psx * 0.5f;
  boxes[o + 3] = pcy + psy * 0.5f;
  scores[(long long)b * NDET + n] = conf[((long long)b * NA + a) * 49 + sp];
}

// ---------------------------------------------------------------------------
// Per-image NMS: one workgroup per image. Stable descending sort via O(N^2)
// rank, sequential greedy suppression, then top-100 emission matching
// jax.lax.top_k tie-breaking (suppressed entries keep their boxes, score -1).
// ---------------------------------------------------------------------------
#define TOPK 100
#define CONF_THR 0.5f
#define IOU_THR  0.5f
__global__ __launch_bounds__(512) void nms_kernel(
    const float* __restrict__ boxes, const float* __restrict__ scores,
    float* __restrict__ out_b, float* __restrict__ out_s, float* __restrict__ out_v)
{
  __shared__ float s_un[NDET];
  __shared__ float bx1[NDET], by1[NDET], bx2[NDET], by2[NDET];
  __shared__ float ss[NDET], area[NDET];
  __shared__ int   keep[NDET];
  const int b = blockIdx.x;
  const int t = threadIdx.x;

  float cb0 = 0.f, cb1 = 0.f, cb2 = 0.f, cb3 = 0.f, sv = -1.0f;
  if (t < NDET) {
    long long o = ((long long)b * NDET + t) * 4;
    cb0 = fminf(fmaxf(boxes[o + 0], 0.f), 1.f);
    cb1 = fminf(fmaxf(boxes[o + 1], 0.f), 1.f);
    cb2 = fminf(fmaxf(boxes[o + 2], 0.f), 1.f);
    cb3 = fminf(fmaxf(boxes[o + 3], 0.f), 1.f);
    float sc = scores[(long long)b * NDET + t];
    sv = (sc > CONF_THR) ? sc : -1.0f;
    s_un[t] = sv;
  }
  __syncthreads();
  if (t < NDET) {     // stable descending rank
    int rank = 0;
    for (int j = 0; j < NDET; ++j) {
      float sj = s_un[j];
      rank += (sj > sv) || (sj == sv && j < t);
    }
    bx1[rank] = cb0; by1[rank] = cb1; bx2[rank] = cb2; by2[rank] = cb3;
    ss[rank] = sv;
  }
  __syncthreads();
  if (t < NDET) {
    area[t] = fmaxf(bx2[t] - bx1[t], 0.f) * fmaxf(by2[t] - by1[t], 0.f);
    keep[t] = (ss[t] > CONF_THR) ? 1 : 0;
  }
  __syncthreads();
  for (int i = 0; i < NDET - 1; ++i) {
    if (keep[i]) {
      if (t > i && t < NDET && keep[t]) {
        float lx = fmaxf(bx1[i], bx1[t]), ly = fmaxf(by1[i], by1[t]);
        float rx = fminf(bx2[i], bx2[t]), ry = fminf(by2[i], by2[t]);
        float inter = fmaxf(rx - lx, 0.f) * fmaxf(ry - ly, 0.f);
        float iou = inter / fmaxf(area[i] + area[t] - inter, 1e-9f);
        if (iou > IOU_THR) keep[t] = 0;
      }
    }
    __syncthreads();
  }
  int posK = 0, total = 0;
  if (t < NDET) {
    for (int j = 0; j < NDET; ++j) { total += keep[j]; if (j < t) posK += keep[j]; }
  }
  __syncthreads();
  if (t < NDET) {
    int slot = keep[t] ? posK : (total + (t - posK));
    if (slot < TOPK) {
      long long ob = (long long)b * TOPK + slot;
      if (total > 0) {
        out_b[ob * 4 + 0] = bx1[t]; out_b[ob * 4 + 1] = by1[t];
        out_b[ob * 4 + 2] = bx2[t]; out_b[ob * 4 + 3] = by2[t];
        out_s[ob] = keep[t] ? ss[t] : -1.0f;
        out_v[ob] = keep[t] ? 1.0f : 0.0f;
      } else {   // fallback box
        if (slot == 0) {
          out_b[ob * 4 + 0] = 0.2f; out_b[ob * 4 + 1] = 0.2f;
          out_b[ob * 4 + 2] = 0.8f; out_b[ob * 4 + 3] = 0.8f;
          out_s[ob] = CONF_THR; out_v[ob] = 1.0f;
        } else {
          out_b[ob * 4 + 0] = 0.f; out_b[ob * 4 + 1] = 0.f;
          out_b[ob * 4 + 2] = 0.f; out_b[ob * 4 + 3] = 0.f;
          out_s[ob] = -1.0f; out_v[ob] = 0.0f;
        }
      }
    }
  }
}

// ---------------------------------------------------------------------------
// Host orchestration
// ---------------------------------------------------------------------------
extern "C" void kernel_launch(void* const* d_in, const int* in_sizes, int n_in,
                              void* d_out, int out_size, void* d_ws, size_t ws_size,
                              hipStream_t stream) {
  (void)in_sizes; (void)n_in; (void)out_size; (void)ws_size;

  auto F = [&](int i) -> const float* { return (const float*)d_in[i]; };
  float* ws = (float*)d_ws;

  // --- param leaf indices (pytree order: dict keys sorted, lists in order) ---
  const int L1B0 = 23, L1B1 = 33, L2B0 = 43, L2B1 = 58;
  const int L3B0 = 68, L3B1 = 83, L4B0 = 93, L4B1 = 108;
  const int P_POOL_B = 118, P_POOL_BN = 119, P_POOL_W = 123;
  const int P_SM1_B = 124, P_SM1_W = 125, P_SM2_B = 126, P_SM2_W = 127;
  const int P_STEM_BN = 128, P_STEM_W = 132;

  struct BNp { const float *g, *b, *m, *v; };
  auto bnAt = [&](int i) { BNp r; r.b = F(i); r.g = F(i + 1); r.m = F(i + 2); r.v = F(i + 3); return r; };

  // --- workspace layout (float offsets) ---
  const size_t SLAB_A = 0;                           // 25,690,112 (stem; reused)
  const size_t BUF_B  = 25690112;                    // 6,422,528
  const size_t BUF_T  = BUF_B  + 6422528;
  const size_t BUF_U  = BUF_T  + 6422528;
  const size_t BUF_C3 = BUF_U  + 6422528;            // 3,211,264
  const size_t BUF_SC = BUF_C3 + 3211264;            // 3,211,264
  const size_t BUF_C4 = BUF_SC + 3211264;            // 1,605,632
  const size_t BUF_C5 = BUF_C4 + 1605632;            // 802,816
  const size_t BUF_P5 = BUF_C5 + 802816;             // 401,408
  const size_t BUF_P4 = BUF_P5 + 401408;             // 1,605,632
  const size_t BUF_P4S= BUF_P4 + 1605632;
  const size_t BUF_F14= BUF_P4S+ 1605632;
  const size_t BUF_T14= BUF_F14+ 1605632;
  const size_t BUF_F7 = BUF_T14+ 1605632;            // 401,408
  const size_t BUF_H1 = BUF_F7 + 401408;
  const size_t BUF_H2 = BUF_H1 + 401408;
  const size_t BUF_H3 = BUF_H2 + 401408;
  const size_t BUF_BB = BUF_H3 + 401408;             // 56,448
  const size_t BUF_CL = BUF_BB + 56448;              // 14,112
  const size_t BUF_BX = BUF_CL + 14112;              // 56,448
  const size_t BUF_CF = BUF_BX + 56448;              // 14,112
  const size_t BUF_P3  = SLAB_A;                     // reuse stem slab
  const size_t BUF_P3S = SLAB_A + 6422528;
  const size_t BUF_T28 = SLAB_A + 12845056;

  auto conv = [&](const float* X, const float* Wt, const float* bias,
                  const float* g, const float* bb, const float* mm, const float* vv,
                  const float* res, float* Y,
                  int N, int Ci, int Hi, int Wi, int Co, int Kh, int Kw,
                  int s, int p, int flags) {
    int Ho = (Hi + 2 * p - Kh) / s + 1;
    int Wo = (Wi + 2 * p - Kw) / s + 1;
    long long Ntot = (long long)N * Ho * Wo;
    dim3 grid((unsigned)((Ntot + 127) / 128), (unsigned)((Co + 63) / 64));
    conv_wmma_f16<<<grid, 256, 0, stream>>>(X, Wt, bias, g, bb, mm, vv, res, Y,
                                            N, Ci, Hi, Wi, Co, Ho, Wo, Kh, Kw, s, p, flags);
  };
  auto maxpool = [&](const float* X, float* Y, int N, int C, int Hi, int Wi,
                     int win, int s, int p) {
    int Ho = (Hi + 2 * p - win) / s + 1;
    int Wo = (Wi + 2 * p - win) / s + 1;
    long long tot = (long long)N * C * Ho * Wo;
    maxpool_kernel<<<(int)((tot + 255) / 256), 256, 0, stream>>>(X, Y, N, C, Hi, Wi, Ho, Wo, win, s, p);
  };
  auto block = [&](const float* in, int base, bool down, int Ci, int Co, int s,
                   int Hi, int Wi, float* t, float* sc, float* out) {
    BNp bn1 = bnAt(base), bn2 = bnAt(base + 4);
    conv(in, F(base + 8), nullptr, bn1.g, bn1.b, bn1.m, bn1.v, nullptr, t,
         32, Ci, Hi, Wi, Co, 3, 3, s, 1, EPI_BN | EPI_RELU);
    int Ho = (Hi + 2 - 3) / s + 1, Wo = (Wi + 2 - 3) / s + 1;
    const float* resp = in;
    if (down) {
      BNp dbn = bnAt(base + 10);
      conv(in, F(base + 14), nullptr, dbn.g, dbn.b, dbn.m, dbn.v, nullptr, sc,
           32, Ci, Hi, Wi, Co, 1, 1, s, 0, EPI_BN);
      resp = sc;
    }
    conv(t, F(base + 9), nullptr, bn2.g, bn2.b, bn2.m, bn2.v, resp, out,
         32, Co, Ho, Wo, Co, 3, 3, 1, 1, EPI_BN | EPI_RES | EPI_RELU);
  };

  const float* x = F(0);

  // --- ResNet18 stem ---
  BNp sbn = bnAt(P_STEM_BN);
  conv(x, F(P_STEM_W), nullptr, sbn.g, sbn.b, sbn.m, sbn.v, nullptr, ws + SLAB_A,
       32, 3, 224, 224, 64, 7, 7, 2, 3, EPI_BN | EPI_RELU);            // 112x112
  maxpool(ws + SLAB_A, ws + BUF_B, 32, 64, 112, 112, 3, 2, 1);         // 56x56

  // --- layers ---
  block(ws + BUF_B, L1B0, false, 64, 64, 1, 56, 56, ws + BUF_T, ws + BUF_SC, ws + BUF_U);
  block(ws + BUF_U, L1B1, false, 64, 64, 1, 56, 56, ws + BUF_T, ws + BUF_SC, ws + BUF_B);   // c2
  block(ws + BUF_B, L2B0, true, 64, 128, 2, 56, 56, ws + BUF_T, ws + BUF_SC, ws + BUF_U);
  block(ws + BUF_U, L2B1, false, 128, 128, 1, 28, 28, ws + BUF_T, ws + BUF_SC, ws + BUF_C3); // c3
  block(ws + BUF_C3, L3B0, true, 128, 256, 2, 28, 28, ws + BUF_T, ws + BUF_SC, ws + BUF_U);
  block(ws + BUF_U, L3B1, false, 256, 256, 1, 14, 14, ws + BUF_T, ws + BUF_SC, ws + BUF_C4); // c4
  block(ws + BUF_C4, L4B0, true, 256, 512, 2, 14, 14, ws + BUF_T, ws + BUF_SC, ws + BUF_U);
  block(ws + BUF_U, L4B1, false, 512, 512, 1, 7, 7, ws + BUF_T, ws + BUF_SC, ws + BUF_C5);   // c5

  // --- FPN ---
  conv(ws + BUF_C5, F(18), F(17), nullptr, nullptr, nullptr, nullptr, nullptr, ws + BUF_P5,
       32, 512, 7, 7, 256, 1, 1, 1, 0, EPI_BIAS);                                      // p5
  conv(ws + BUF_C4, F(20), F(19), nullptr, nullptr, nullptr, nullptr, nullptr, ws + BUF_P4,
       32, 256, 14, 14, 256, 1, 1, 1, 0, EPI_BIAS);
  {
    long long tot = 32LL * 256 * 14 * 14;
    upsample_add_kernel<<<(int)((tot + 255) / 256), 256, 0, stream>>>(ws + BUF_P4, ws + BUF_P5, 32, 256, 14, 14, 7, 7);
  }
  conv(ws + BUF_P4, F(P_SM1_W), F(P_SM1_B), nullptr, nullptr, nullptr, nullptr, nullptr, ws + BUF_P4S,
       32, 256, 14, 14, 256, 3, 3, 1, 1, EPI_BIAS);                                    // p4
  conv(ws + BUF_C3, F(22), F(21), nullptr, nullptr, nullptr, nullptr, nullptr, ws + BUF_P3,
       32, 128, 28, 28, 256, 1, 1, 1, 0, EPI_BIAS);
  {
    long long tot = 32LL * 256 * 28 * 28;
    upsample_add_kernel<<<(int)((tot + 255) / 256), 256, 0, stream>>>(ws + BUF_P3, ws + BUF_P4S, 32, 256, 28, 28, 14, 14);
  }
  conv(ws + BUF_P3, F(P_SM2_W), F(P_SM2_B), nullptr, nullptr, nullptr, nullptr, nullptr, ws + BUF_P3S,
       32, 256, 28, 28, 256, 3, 3, 1, 1, EPI_BIAS);                                    // p3

  // --- pool 28 -> 14 -> 7 (shared weights) ---
  BNp pbn = bnAt(P_POOL_BN);
  conv(ws + BUF_P3S, F(P_POOL_W), F(P_POOL_B), pbn.g, pbn.b, pbn.m, pbn.v, nullptr, ws + BUF_T28,
       32, 256, 28, 28, 256, 3, 3, 1, 1, EPI_BIAS | EPI_BN | EPI_RELU);
  maxpool(ws + BUF_T28, ws + BUF_F14, 32, 256, 28, 28, 2, 2, 0);
  conv(ws + BUF_F14, F(P_POOL_W), F(P_POOL_B), pbn.g, pbn.b, pbn.m, pbn.v, nullptr, ws + BUF_T14,
       32, 256, 14, 14, 256, 3, 3, 1, 1, EPI_BIAS | EPI_BN | EPI_RELU);
  maxpool(ws + BUF_T14, ws + BUF_F7, 32, 256, 14, 14, 2, 2, 0);

  // --- detection head ---
  BNp h1bn = bnAt(5), h2bn = bnAt(9);
  conv(ws + BUF_F7, F(14), F(13), h1bn.g, h1bn.b, h1bn.m, h1bn.v, nullptr, ws + BUF_H1,
       32, 256, 7, 7, 256, 3, 3, 1, 1, EPI_BIAS | EPI_BN | EPI_RELU);
  conv(ws + BUF_H1, F(16), F(15), h2bn.g, h2bn.b, h2bn.m, h2bn.v, nullptr, ws + BUF_H2,
       32, 256, 7, 7, 256, 3, 3, 1, 1, EPI_BIAS | EPI_BN | EPI_RELU);
  {
    long long tot = 32LL * 256 * 7 * 7;
    add_kernel<<<(int)((tot + 255) / 256), 256, 0, stream>>>(ws + BUF_H2, ws + BUF_F7, ws + BUF_H3, tot);
  }
  conv(ws + BUF_H3, F(2), F(1), nullptr, nullptr, nullptr, nullptr, nullptr, ws + BUF_BB,
       32, 256, 7, 7, 36, 3, 3, 1, 1, EPI_BIAS);                                       // bbox
  conv(ws + BUF_H3, F(4), F(3), nullptr, nullptr, nullptr, nullptr, nullptr, ws + BUF_CL,
       32, 256, 7, 7, 9, 3, 3, 1, 1, EPI_BIAS | EPI_SIGMOID);                          // conf

  // --- decode + NMS ---
  decode_kernel<<<(32 * NDET + 255) / 256, 256, 0, stream>>>(ws + BUF_BB, ws + BUF_CL,
                                                             ws + BUF_BX, ws + BUF_CF, 32);
  float* out_b = (float*)d_out;              // [32,100,4]
  float* out_s = out_b + 32 * TOPK * 4;      // [32,100]
  float* out_v = out_s + 32 * TOPK;          // [32,100]
  nms_kernel<<<32, 512, 0, stream>>>(ws + BUF_BX, ws + BUF_CF, out_b, out_s, out_v);
}
